// EncodeProcessDecode_51058571215552
// MI455X (gfx1250) — matrix operand order repaired and checked by reading, not compiled, not run on hardware
//
#include <hip/hip_runtime.h>

// ---------------------------------------------------------------------------
// Types / WMMA helper
// ---------------------------------------------------------------------------
typedef _Float16 v16h __attribute__((ext_vector_type(16)));
typedef float    v8f  __attribute__((ext_vector_type(8)));

__device__ __forceinline__ v8f wmma16(v16h a, v16h b, v8f c) {
  // D = A(16x32 f16) * B(32x16 f16) + C(16x16 f32)
  return __builtin_amdgcn_wmma_f32_16x16x32_f16(
      /*neg_a=*/false, a, /*neg_b=*/false, b,
      /*c_mod=*/(short)0, c, /*reuse_a=*/false, /*reuse_b=*/false);
}

#define LAT 128
#define EPSLN 1e-5f

// ---------------------------------------------------------------------------
// Weight convert: W [S][IN][OUT] f32 -> WT [S][OUT][IN] f16 (K-contiguous B)
// ---------------------------------------------------------------------------
__global__ void convert_wt_kernel(const float* __restrict__ W, _Float16* __restrict__ WT,
                                  int S, int IN, int OUT, int total) {
  int idx = blockIdx.x * 256 + threadIdx.x;
  if (idx >= total) return;
  int per = IN * OUT;
  int s = idx / per, rem = idx - s * per;
  int i = rem / OUT, o = rem - i * OUT;
  WT[(size_t)s * per + (size_t)o * IN + i] = (_Float16)W[idx];
}

__global__ void zero_f32_kernel(float* __restrict__ p, long long n) {
  long long i = (long long)blockIdx.x * 256 + threadIdx.x;
  if (i < n) p[i] = 0.0f;
}

// ---------------------------------------------------------------------------
// Encoder MLP (fp32 VALU, tiny): out[n] = LN(x W0 + b0) chain, 1 node/block
// ---------------------------------------------------------------------------
__global__ __launch_bounds__(128) void encode_mlp_ln_kernel(
    const float* __restrict__ x, int in_dim,
    const float* __restrict__ W0, const float* __restrict__ b0,
    const float* __restrict__ W1, const float* __restrict__ b1,
    const float* __restrict__ W2, const float* __restrict__ b2,
    const float* __restrict__ g,  const float* __restrict__ beta,
    float* __restrict__ outbuf) {
  __shared__ float xs[16];
  __shared__ float h[LAT];
  __shared__ float s1[LAT], s2[LAT];
  const int t = threadIdx.x, n = blockIdx.x;
  if (t < in_dim) xs[t] = x[(size_t)n * in_dim + t];
  __syncthreads();
  float a0 = b0[t];
  for (int k = 0; k < in_dim; ++k) a0 += xs[k] * W0[k * LAT + t];
  h[t] = a0; __syncthreads();
  float a1 = b1[t];
  for (int k = 0; k < LAT; ++k) a1 += h[k] * W1[k * LAT + t];
  __syncthreads(); h[t] = a1; __syncthreads();
  float a2 = b2[t];
  for (int k = 0; k < LAT; ++k) a2 += h[k] * W2[k * LAT + t];
  s1[t] = a2; s2[t] = a2 * a2; __syncthreads();
  for (int off = 64; off > 0; off >>= 1) {
    if (t < off) { s1[t] += s1[t + off]; s2[t] += s2[t + off]; }
    __syncthreads();
  }
  const float mu  = s1[0] * (1.0f / LAT);
  const float var = s2[0] * (1.0f / LAT) - mu * mu;
  const float rs  = rsqrtf(var + EPSLN);
  outbuf[(size_t)n * LAT + t] = (a2 - mu) * rs * g[t] + beta[t];
}

// ---------------------------------------------------------------------------
// Decoder MLP (fp32 VALU, tiny): 128 -> 128 -> 128 -> 3, no LN
// ---------------------------------------------------------------------------
__global__ __launch_bounds__(128) void decode_mlp_kernel(
    const float* __restrict__ v,
    const float* __restrict__ W0, const float* __restrict__ b0,
    const float* __restrict__ W1, const float* __restrict__ b1,
    const float* __restrict__ W2, const float* __restrict__ b2,
    float* __restrict__ out) {
  __shared__ float h[LAT];
  __shared__ float h2[LAT];
  const int t = threadIdx.x, n = blockIdx.x;
  const float* xr = v + (size_t)n * LAT;
  float a0 = b0[t];
  for (int k = 0; k < LAT; ++k) a0 += xr[k] * W0[k * LAT + t];
  h[t] = a0; __syncthreads();
  float a1 = b1[t];
  for (int k = 0; k < LAT; ++k) a1 += h[k] * W1[k * LAT + t];
  h2[t] = a1; __syncthreads();
  if (t < 3) {
    float a2 = b2[t];
    for (int k = 0; k < LAT; ++k) a2 += h2[k] * W2[k * 3 + t];
    out[(size_t)n * 3 + t] = a2;
  }
}

// ---------------------------------------------------------------------------
// Edge step: one wave (32 lanes) per 16-edge tile.
//   ef = [v[snd] | v[rcv] | e]  (K=384) -> L0 -> L1 -> L2 -> LN
//   e += e_upd (residual);  agg[rcv] += e_upd (segment sum via f32 atomics)
// ---------------------------------------------------------------------------
__global__ __launch_bounds__(32) void edge_step_kernel(
    const float* __restrict__ v, float* __restrict__ e, float* __restrict__ agg,
    const int* __restrict__ senders, const int* __restrict__ receivers,
    const _Float16* __restrict__ W0T, const _Float16* __restrict__ W1T,
    const _Float16* __restrict__ W2T,
    const float* __restrict__ b0, const float* __restrict__ b1,
    const float* __restrict__ b2,
    const float* __restrict__ g, const float* __restrict__ beta, int E) {
  __shared__ float hbuf[16 * LAT];
  const int lane = threadIdx.x;
  const int r  = lane & 15;      // row within 16-row tile (A-matrix row)
  const int hf = lane >> 4;      // half-wave select
  const int base = blockIdx.x * 16;
  int row_g = base + r; if (row_g >= E) row_g = E - 1;
  const int snd = senders[row_g];
  const int rcv = receivers[row_g];
  const float* vs = v + (size_t)snd * LAT;
  const float* vr = v + (size_t)rcv * LAT;
  const float* es = e + (size_t)row_g * LAT;

  v8f c[8];
#pragma unroll
  for (int nt = 0; nt < 8; ++nt)
#pragma unroll
    for (int j = 0; j < 8; ++j) c[nt][j] = 0.0f;

  // ---------- layer 0: K = 384 (gather-fused A build) ----------
  for (int ks = 0; ks < 12; ++ks) {
    v16h a;
#pragma unroll
    for (int ch = 0; ch < 2; ++ch) {
      const int kk = ks * 32 + hf * 16 + ch * 8;   // 8-chunk stays inside one segment
      const float* src = (kk < 128) ? (vs + kk)
                       : (kk < 256) ? (vr + (kk - 128))
                                    : (es + (kk - 256));
#pragma unroll
      for (int i = 0; i < 8; ++i) a[ch * 8 + i] = (_Float16)src[i];
    }
    const int kb = ks * 32 + hf * 16;
#pragma unroll
    for (int nt = 0; nt < 8; ++nt) {
      const int col = nt * 16 + r;
      v16h bv = *(const v16h*)(W0T + (size_t)col * 384 + kb);
      c[nt] = wmma16(a, bv, c[nt]);
    }
  }
#pragma unroll
  for (int nt = 0; nt < 8; ++nt) {
    const float bb = b0[nt * 16 + r];
#pragma unroll
    for (int j = 0; j < 8; ++j)
      hbuf[(hf * 8 + j) * LAT + nt * 16 + r] = c[nt][j] + bb;  // C-layout -> LDS
  }
  __syncthreads();

  // ---------- layers 1 & 2: K = 128, A from LDS transpose ----------
  for (int layer = 0; layer < 2; ++layer) {
    const _Float16* WT = layer ? W2T : W1T;
    const float* bp = layer ? b2 : b1;
#pragma unroll
    for (int nt = 0; nt < 8; ++nt)
#pragma unroll
      for (int j = 0; j < 8; ++j) c[nt][j] = 0.0f;
    for (int ks = 0; ks < 4; ++ks) {
      v16h a;
#pragma unroll
      for (int ch = 0; ch < 2; ++ch) {
        const int kk = ks * 32 + hf * 16 + ch * 8;
#pragma unroll
        for (int i = 0; i < 8; ++i) a[ch * 8 + i] = (_Float16)hbuf[r * LAT + kk + i];
      }
      const int kb = ks * 32 + hf * 16;
#pragma unroll
      for (int nt = 0; nt < 8; ++nt) {
        const int col = nt * 16 + r;
        v16h bv = *(const v16h*)(WT + (size_t)col * LAT + kb);
        c[nt] = wmma16(a, bv, c[nt]);
      }
    }
    __syncthreads();
#pragma unroll
    for (int nt = 0; nt < 8; ++nt) {
      const float bb = bp[nt * 16 + r];
#pragma unroll
      for (int j = 0; j < 8; ++j)
        hbuf[(hf * 8 + j) * LAT + nt * 16 + r] = c[nt][j] + bb;
    }
    __syncthreads();
  }

  // ---------- LayerNorm row stats (half-wave) ----------
  float mu = 0.0f, rs = 0.0f;
  if (lane < 16) {
    float s = 0.0f, s2 = 0.0f;
    for (int k = 0; k < LAT; ++k) { float xv = hbuf[lane * LAT + k]; s += xv; s2 += xv * xv; }
    mu = s * (1.0f / LAT);
    const float var = s2 * (1.0f / LAT) - mu * mu;
    rs = rsqrtf(var + EPSLN);
  }

  // ---------- epilogue: e residual + scatter-add into agg ----------
  for (int i = 0; i < 64; ++i) {
    const int idx = i * 32 + lane;          // coalesced over cols
    const int row = idx >> 7;
    const int col = idx & 127;
    const float m  = __shfl(mu, row);
    const float sc = __shfl(rs, row);
    const int   rv = __shfl(rcv, row);
    if (base + row < E) {
      const float val = (hbuf[row * LAT + col] - m) * sc * g[col] + beta[col];
      const size_t eo = (size_t)(base + row) * LAT + col;
      e[eo] += val;                          // edge residual
      __hip_atomic_fetch_add(&agg[(size_t)rv * LAT + col], val,
                             __ATOMIC_RELAXED, __HIP_MEMORY_SCOPE_AGENT);
    }
  }
}

// ---------------------------------------------------------------------------
// Node step: one wave per 16-node tile.  A = [v | agg] (K=256) -> MLP -> LN
//   v += v_upd  (residual)
// ---------------------------------------------------------------------------
__global__ __launch_bounds__(32) void node_step_kernel(
    float* __restrict__ v, const float* __restrict__ agg,
    const _Float16* __restrict__ W0T, const _Float16* __restrict__ W1T,
    const _Float16* __restrict__ W2T,
    const float* __restrict__ b0, const float* __restrict__ b1,
    const float* __restrict__ b2,
    const float* __restrict__ g, const float* __restrict__ beta, int N) {
  __shared__ float hbuf[16 * LAT];
  const int lane = threadIdx.x;
  const int r  = lane & 15;
  const int hf = lane >> 4;
  const int base = blockIdx.x * 16;
  int row_g = base + r; if (row_g >= N) row_g = N - 1;
  const float* vs = v   + (size_t)row_g * LAT;
  const float* as = agg + (size_t)row_g * LAT;

  v8f c[8];
#pragma unroll
  for (int nt = 0; nt < 8; ++nt)
#pragma unroll
    for (int j = 0; j < 8; ++j) c[nt][j] = 0.0f;

  // ---------- layer 0: K = 256 ----------
  for (int ks = 0; ks < 8; ++ks) {
    v16h a;
#pragma unroll
    for (int ch = 0; ch < 2; ++ch) {
      const int kk = ks * 32 + hf * 16 + ch * 8;
      const float* src = (kk < 128) ? (vs + kk) : (as + (kk - 128));
#pragma unroll
      for (int i = 0; i < 8; ++i) a[ch * 8 + i] = (_Float16)src[i];
    }
    const int kb = ks * 32 + hf * 16;
#pragma unroll
    for (int nt = 0; nt < 8; ++nt) {
      const int col = nt * 16 + r;
      v16h bv = *(const v16h*)(W0T + (size_t)col * 256 + kb);
      c[nt] = wmma16(a, bv, c[nt]);
    }
  }
#pragma unroll
  for (int nt = 0; nt < 8; ++nt) {
    const float bb = b0[nt * 16 + r];
#pragma unroll
    for (int j = 0; j < 8; ++j)
      hbuf[(hf * 8 + j) * LAT + nt * 16 + r] = c[nt][j] + bb;
  }
  __syncthreads();

  // ---------- layers 1 & 2 ----------
  for (int layer = 0; layer < 2; ++layer) {
    const _Float16* WT = layer ? W2T : W1T;
    const float* bp = layer ? b2 : b1;
#pragma unroll
    for (int nt = 0; nt < 8; ++nt)
#pragma unroll
      for (int j = 0; j < 8; ++j) c[nt][j] = 0.0f;
    for (int ks = 0; ks < 4; ++ks) {
      v16h a;
#pragma unroll
      for (int ch = 0; ch < 2; ++ch) {
        const int kk = ks * 32 + hf * 16 + ch * 8;
#pragma unroll
        for (int i = 0; i < 8; ++i) a[ch * 8 + i] = (_Float16)hbuf[r * LAT + kk + i];
      }
      const int kb = ks * 32 + hf * 16;
#pragma unroll
      for (int nt = 0; nt < 8; ++nt) {
        const int col = nt * 16 + r;
        v16h bv = *(const v16h*)(WT + (size_t)col * LAT + kb);
        c[nt] = wmma16(a, bv, c[nt]);
      }
    }
    __syncthreads();
#pragma unroll
    for (int nt = 0; nt < 8; ++nt) {
      const float bb = bp[nt * 16 + r];
#pragma unroll
      for (int j = 0; j < 8; ++j)
        hbuf[(hf * 8 + j) * LAT + nt * 16 + r] = c[nt][j] + bb;
    }
    __syncthreads();
  }

  // ---------- LayerNorm + residual write-back ----------
  float mu = 0.0f, rs = 0.0f;
  if (lane < 16) {
    float s = 0.0f, s2 = 0.0f;
    for (int k = 0; k < LAT; ++k) { float xv = hbuf[lane * LAT + k]; s += xv; s2 += xv * xv; }
    mu = s * (1.0f / LAT);
    const float var = s2 * (1.0f / LAT) - mu * mu;
    rs = rsqrtf(var + EPSLN);
  }
  for (int i = 0; i < 64; ++i) {
    const int idx = i * 32 + lane;
    const int row = idx >> 7;
    const int col = idx & 127;
    const float m  = __shfl(mu, row);
    const float sc = __shfl(rs, row);
    if (base + row < N) {
      const float val = (hbuf[row * LAT + col] - m) * sc * g[col] + beta[col];
      v[(size_t)(base + row) * LAT + col] += val;   // v_new = mlp + v
    }
  }
}

// ---------------------------------------------------------------------------
// Host-side orchestration
// ---------------------------------------------------------------------------
extern "C" void kernel_launch(void* const* d_in, const int* in_sizes, int n_in,
                              void* d_out, int out_size, void* d_ws, size_t ws_size,
                              hipStream_t stream) {
  // ---- inputs (setup_inputs dict insertion order, nested dicts flattened) ----
  const float* node_features = (const float*)d_in[0];
  const float* edge_features = (const float*)d_in[1];
  const int*   senders       = (const int*)d_in[2];
  const int*   receivers     = (const int*)d_in[3];
  // enc_node: W0 b0 W1 b1 W2 b2 g beta
  const float* enW0 = (const float*)d_in[4];  const float* enb0 = (const float*)d_in[5];
  const float* enW1 = (const float*)d_in[6];  const float* enb1 = (const float*)d_in[7];
  const float* enW2 = (const float*)d_in[8];  const float* enb2 = (const float*)d_in[9];
  const float* eng  = (const float*)d_in[10]; const float* enbt = (const float*)d_in[11];
  // enc_edge
  const float* eeW0 = (const float*)d_in[12]; const float* eeb0 = (const float*)d_in[13];
  const float* eeW1 = (const float*)d_in[14]; const float* eeb1 = (const float*)d_in[15];
  const float* eeW2 = (const float*)d_in[16]; const float* eeb2 = (const float*)d_in[17];
  const float* eeg  = (const float*)d_in[18]; const float* eebt = (const float*)d_in[19];
  // blk_edge (stacked S=15)
  const float* beW0 = (const float*)d_in[20]; const float* beb0 = (const float*)d_in[21];
  const float* beW1 = (const float*)d_in[22]; const float* beb1 = (const float*)d_in[23];
  const float* beW2 = (const float*)d_in[24]; const float* beb2 = (const float*)d_in[25];
  const float* beg  = (const float*)d_in[26]; const float* bebt = (const float*)d_in[27];
  // blk_node (stacked S=15)
  const float* bnW0 = (const float*)d_in[28]; const float* bnb0 = (const float*)d_in[29];
  const float* bnW1 = (const float*)d_in[30]; const float* bnb1 = (const float*)d_in[31];
  const float* bnW2 = (const float*)d_in[32]; const float* bnb2 = (const float*)d_in[33];
  const float* bng  = (const float*)d_in[34]; const float* bnbt = (const float*)d_in[35];
  // dec (no LN)
  const float* dW0 = (const float*)d_in[36]; const float* db0 = (const float*)d_in[37];
  const float* dW1 = (const float*)d_in[38]; const float* db1 = (const float*)d_in[39];
  const float* dW2 = (const float*)d_in[40]; const float* db2 = (const float*)d_in[41];

  float* out = (float*)d_out;
  const int N = in_sizes[0] / 12;
  const int E = in_sizes[2];
  const int S = 15;

  // ---- workspace carve ----
  char* p = (char*)d_ws;
  auto carve = [&](size_t bytes) -> char* {
    char* q = p; p += (bytes + 255) & ~(size_t)255; return q;
  };
  float* v   = (float*)carve((size_t)N * LAT * sizeof(float));
  float* agg = (float*)carve((size_t)N * LAT * sizeof(float));
  float* e   = (float*)carve((size_t)E * LAT * sizeof(float));
  _Float16* W0Te = (_Float16*)carve((size_t)S * 128 * 384 * sizeof(_Float16));
  _Float16* W1Te = (_Float16*)carve((size_t)S * 128 * 128 * sizeof(_Float16));
  _Float16* W2Te = (_Float16*)carve((size_t)S * 128 * 128 * sizeof(_Float16));
  _Float16* W0Tn = (_Float16*)carve((size_t)S * 128 * 256 * sizeof(_Float16));
  _Float16* W1Tn = (_Float16*)carve((size_t)S * 128 * 128 * sizeof(_Float16));
  _Float16* W2Tn = (_Float16*)carve((size_t)S * 128 * 128 * sizeof(_Float16));

  // ---- convert stacked weights to f16, [S][OUT][IN] (B K-contiguous) ----
  auto cvt = [&](const float* W, _Float16* WT, int IN, int OUT) {
    int total = S * IN * OUT;
    convert_wt_kernel<<<(total + 255) / 256, 256, 0, stream>>>(W, WT, S, IN, OUT, total);
  };
  cvt(beW0, W0Te, 384, 128); cvt(beW1, W1Te, 128, 128); cvt(beW2, W2Te, 128, 128);
  cvt(bnW0, W0Tn, 256, 128); cvt(bnW1, W1Tn, 128, 128); cvt(bnW2, W2Tn, 128, 128);

  // ---- encode (fp32) ----
  encode_mlp_ln_kernel<<<N, 128, 0, stream>>>(node_features, 12,
      enW0, enb0, enW1, enb1, enW2, enb2, eng, enbt, v);
  encode_mlp_ln_kernel<<<E, 128, 0, stream>>>(edge_features, 7,
      eeW0, eeb0, eeW1, eeb1, eeW2, eeb2, eeg, eebt, e);

  // ---- process: 15 GraphNet blocks (WMMA) ----
  const int etiles = (E + 15) / 16;
  const int ntiles = (N + 15) / 16;
  const long long aggN = (long long)N * LAT;
  for (int s = 0; s < S; ++s) {
    zero_f32_kernel<<<(int)((aggN + 255) / 256), 256, 0, stream>>>(agg, aggN);
    edge_step_kernel<<<etiles, 32, 0, stream>>>(
        v, e, agg, senders, receivers,
        W0Te + (size_t)s * 128 * 384,
        W1Te + (size_t)s * 128 * 128,
        W2Te + (size_t)s * 128 * 128,
        beb0 + s * 128, beb1 + s * 128, beb2 + s * 128,
        beg + s * 128, bebt + s * 128, E);
    node_step_kernel<<<ntiles, 32, 0, stream>>>(
        v, agg,
        W0Tn + (size_t)s * 128 * 256,
        W1Tn + (size_t)s * 128 * 128,
        W2Tn + (size_t)s * 128 * 128,
        bnb0 + s * 128, bnb1 + s * 128, bnb2 + s * 128,
        bng + s * 128, bnbt + s * 128, N);
  }

  // ---- decode (fp32) ----
  decode_mlp_kernel<<<N, 128, 0, stream>>>(v, dW0, db0, dW1, db1, dW2, db2, out);
}
